// AttentionGate_12463995093220
// MI455X (gfx1250) — matrix operand order repaired and checked by reading, not compiled, not run on hardware
//
#include <hip/hip_runtime.h>
#include <hip/hip_bf16.h>
#include <math.h>

#define BB 32
#define SS 2048
#define HH 1024
#define AA 1024
#define SROWS 64                  // 4 row-tiles of 16 per workgroup
#define RT    (SROWS / 16)        // 4
#define LDA (HH + 8)              // padded f16 row stride in LDS (bank-conflict dodge)

typedef __attribute__((ext_vector_type(16))) _Float16 v16h;
typedef __attribute__((ext_vector_type(8)))  _Float16 v8h;
typedef __attribute__((ext_vector_type(4)))  _Float16 v4h;
typedef __attribute__((ext_vector_type(8)))  float    v8f;

// ---------------------------------------------------------------------------
// WMMA helpers (CDNA5 f16 16x16x32, f32 accumulate)
// ---------------------------------------------------------------------------
__device__ __forceinline__ v8f wmma_f16(v16h a, v16h b, v8f c) {
    // 8 args: (neg_a, A, neg_b, B, c_mod, C, reuse_a, reuse_b)
    return __builtin_amdgcn_wmma_f32_16x16x32_f16(false, a, false, b,
                                                  (short)0, c, false, false);
}

// A-matrix 16x32 f16 fragment from LDS (row-major, stride lda halves).
// Contiguous k order used here (B matches): lane l holds row M=l%16; K = 16*(l/16)+0..15
__device__ __forceinline__ v16h load_a_frag_lds(const _Float16* row_base,
                                                int lda, int lane, int kbase) {
    const int m  = lane & 15;
    const int hi = (lane >> 4) & 1;
    const _Float16* p = row_base + m * lda + kbase + hi * 16;
    v8h lo  = *(const v8h*)(p);
    v8h hi8 = *(const v8h*)(p + 8);
    v16h a;
#pragma unroll
    for (int j = 0; j < 8; ++j) { a[j] = lo[j]; a[8 + j] = hi8[j]; }
    return a;
}

// ---------------------------------------------------------------------------
// Kernel 0: fp32 W[R][C] -> f16 transposed WT[C][R] (B-fragment friendly)
// ---------------------------------------------------------------------------
__global__ __launch_bounds__(256) void conv_t_kernel(const float* __restrict__ W,
                                                     _Float16* __restrict__ WT,
                                                     int R, int C) {
    int idx = blockIdx.x * 256 + threadIdx.x;
    int c = idx / R;
    int r = idx - c * R;
    WT[(size_t)c * R + r] = (_Float16)W[(size_t)r * C + c];
}

// ---------------------------------------------------------------------------
// Kernel 1: query = q_state @ Wq + bq   [B,A]  (tiny)
// ---------------------------------------------------------------------------
__global__ __launch_bounds__(256) void qproj_kernel(const float* __restrict__ q,
                                                    const float* __restrict__ Wq,
                                                    const float* __restrict__ bq,
                                                    float* __restrict__ query) {
    int idx = blockIdx.x * 256 + threadIdx.x;   // B*A
    int b = idx >> 10;
    int a = idx & (AA - 1);
    const float* qr = q + (size_t)b * HH;
    float acc = bq[a];
    for (int h = 0; h < HH; ++h) acc = fmaf(qr[h], Wq[(size_t)h * AA + a], acc);
    query[idx] = acc;
}

// ---------------------------------------------------------------------------
// Kernel 2: fused  keys GEMM -> tanh -> gate GEMM -> sigmoid*align -> scores
// One workgroup = (batch b, 64 rows of S). 8 waves; wave w owns N columns
// [w*128, w*128+128) as 8 tiles of 16. B-fragments are loaded ONCE per k-step
// and reused across all 4 row-tiles (weights stream exactly once per WG).
// ---------------------------------------------------------------------------
__global__ __launch_bounds__(256, 1) void fused_attn_kernel(
    const float*    __restrict__ seq,    // [B,S,H]
    const float*    __restrict__ query,  // [B,A]
    const _Float16* __restrict__ WkT,    // [A][H] f16
    const float*    __restrict__ bk,     // [A]
    const _Float16* __restrict__ WgT,    // [A][A] f16 (out-col major)
    const float*    __restrict__ bg,     // [A]
    const float*    __restrict__ Ws,     // [A]
    const float*    __restrict__ bs,     // [1]
    float*          __restrict__ scores) // [B,S]
{
    __shared__ _Float16 lds_seq[SROWS * LDA];    // seq tile f16  (~129 KB)
    __shared__ _Float16 lds_align[SROWS * LDA];  // tanh(align)   (~129 KB)
    __shared__ float    lds_sc[SROWS];           // per-row score accumulators

    const int tid    = threadIdx.x;
    const int lane   = tid & 31;
    const int wave   = tid >> 5;
    const int n_lane = lane & 15;
    const int hi     = lane >> 4;

    const int nblk = SS / SROWS;                 // 32 S-tiles per batch
    const int b    = blockIdx.x / nblk;
    const int s0   = (blockIdx.x - b * nblk) * SROWS;

    // ---- stage seq tile: fp32 global -> f16 LDS ----
    const float* seqb = seq + ((size_t)b * SS + s0) * HH;
    for (int i = tid; i < SROWS * HH / 4; i += 256) {
        float4 v = ((const float4*)seqb)[i];
        int l   = i * 4;
        int row = l >> 10;
        int col = l & (HH - 1);
        v4h h;
        h[0] = (_Float16)v.x; h[1] = (_Float16)v.y;
        h[2] = (_Float16)v.z; h[3] = (_Float16)v.w;
        *(v4h*)&lds_seq[row * LDA + col] = h;
    }
    if (tid < SROWS) lds_sc[tid] = 0.0f;
    __syncthreads();

    // ---- GEMM1: keys = seq @ Wk ; align = tanh(keys + query + bk) ----
    for (int nn = 0; nn < 8; ++nn) {
        const int n0 = (wave * 8 + nn) * 16;
        const _Float16* bptr = WkT + (size_t)(n0 + n_lane) * HH + hi * 16;
        v8f acc[RT];
#pragma unroll
        for (int rt = 0; rt < RT; ++rt) acc[rt] = (v8f){};
        for (int kk = 0; kk < HH; kk += 32) {
            __builtin_prefetch(bptr + kk + 256, 0, 1);
            v16h bfr = *(const v16h*)(bptr + kk);      // loaded once...
#pragma unroll
            for (int rt = 0; rt < RT; ++rt) {          // ...reused 4x
                v16h a = load_a_frag_lds(lds_seq + rt * 16 * LDA, LDA, lane, kk);
                acc[rt] = wmma_f16(a, bfr, acc[rt]);
            }
        }
        const int nCol = n0 + n_lane;
        const float addv = query[(size_t)b * AA + nCol] + bk[nCol];
#pragma unroll
        for (int rt = 0; rt < RT; ++rt) {
#pragma unroll
            for (int r = 0; r < 8; ++r) {
                int row = rt * 16 + r + 8 * hi;
                float v = tanhf(acc[rt][r] + addv);
                lds_align[row * LDA + nCol] = (_Float16)v;
            }
        }
    }
    __syncthreads();

    // ---- GEMM2: logits = align @ Wg ; score += align*sigmoid(logits+bg)*Ws ----
    float sp[RT][8];
#pragma unroll
    for (int rt = 0; rt < RT; ++rt)
#pragma unroll
        for (int r = 0; r < 8; ++r) sp[rt][r] = 0.0f;

    for (int nn = 0; nn < 8; ++nn) {
        const int n0 = (wave * 8 + nn) * 16;
        const _Float16* bptr = WgT + (size_t)(n0 + n_lane) * AA + hi * 16;
        v8f acc[RT];
#pragma unroll
        for (int rt = 0; rt < RT; ++rt) acc[rt] = (v8f){};
        for (int kk = 0; kk < AA; kk += 32) {
            __builtin_prefetch(bptr + kk + 256, 0, 1);
            v16h bfr = *(const v16h*)(bptr + kk);
#pragma unroll
            for (int rt = 0; rt < RT; ++rt) {
                v16h a = load_a_frag_lds(lds_align + rt * 16 * LDA, LDA, lane, kk);
                acc[rt] = wmma_f16(a, bfr, acc[rt]);
            }
        }
        const int nCol = n0 + n_lane;
        const float bgv = bg[nCol];
        const float wsv = Ws[nCol];
#pragma unroll
        for (int rt = 0; rt < RT; ++rt) {
#pragma unroll
            for (int r = 0; r < 8; ++r) {
                int row  = rt * 16 + r + 8 * hi;
                float av = (float)lds_align[row * LDA + nCol];
                float g  = 1.0f / (1.0f + expf(-(acc[rt][r] + bgv)));
                sp[rt][r] += av * g * wsv;
            }
        }
    }
#pragma unroll
    for (int rt = 0; rt < RT; ++rt)
#pragma unroll
        for (int r = 0; r < 8; ++r) {
            int row = rt * 16 + r + 8 * hi;
            atomicAdd(&lds_sc[row], sp[rt][r]);        // ds_add_f32
        }
    __syncthreads();
    if (tid < SROWS)
        scores[(size_t)b * SS + s0 + tid] = lds_sc[tid] + bs[0];
}

// ---------------------------------------------------------------------------
// Kernel 3: softmax over S per batch -> weights (second output)
// ---------------------------------------------------------------------------
__global__ __launch_bounds__(256) void softmax_kernel(const float* __restrict__ scores,
                                                      float* __restrict__ weights) {
    __shared__ float red[256];
    const int b = blockIdx.x, t = threadIdx.x;
    const float* sc = scores + (size_t)b * SS;
    float m = -1e30f;
    for (int i = t; i < SS; i += 256) m = fmaxf(m, sc[i]);
    red[t] = m; __syncthreads();
    for (int o = 128; o > 0; o >>= 1) {
        if (t < o) red[t] = fmaxf(red[t], red[t + o]);
        __syncthreads();
    }
    const float gmax = red[0];
    __syncthreads();
    float s = 0.0f;
    for (int i = t; i < SS; i += 256) s += expf(sc[i] - gmax);
    red[t] = s; __syncthreads();
    for (int o = 128; o > 0; o >>= 1) {
        if (t < o) red[t] += red[t + o];
        __syncthreads();
    }
    const float inv = 1.0f / red[0];
    for (int i = t; i < SS; i += 256)
        weights[(size_t)b * SS + i] = expf(sc[i] - gmax) * inv;
}

// ---------------------------------------------------------------------------
// Kernel 4: context[b,h] = sum_s weights[b,s] * seq[b,s,h]  (BW-bound)
// ---------------------------------------------------------------------------
__global__ __launch_bounds__(256) void context_kernel(const float* __restrict__ weights,
                                                      const float* __restrict__ seq,
                                                      float* __restrict__ context) {
    __shared__ float wsh[256];
    const int b = blockIdx.x >> 2;                      // H/256 = 4 chunks
    const int h = ((blockIdx.x & 3) << 8) + threadIdx.x;
    const float* wp = weights + (size_t)b * SS;
    const float* sp = seq + (size_t)b * SS * HH + h;
    float acc = 0.0f;
    for (int sb = 0; sb < SS; sb += 256) {
        __syncthreads();
        wsh[threadIdx.x] = wp[sb + threadIdx.x];
        __syncthreads();
        for (int j = 0; j < 256; ++j)
            acc = fmaf(wsh[j], sp[(size_t)(sb + j) * HH], acc);
    }
    context[(size_t)b * HH + h] = acc;
}

// ---------------------------------------------------------------------------
// Kernel 5: pre = [context,q] @ Wo + bo + q ; LayerNorm -> first output
// ---------------------------------------------------------------------------
__global__ __launch_bounds__(256) void out_ln_kernel(const float* __restrict__ context,
                                                     const float* __restrict__ qst,
                                                     const float* __restrict__ Wo,
                                                     const float* __restrict__ bo,
                                                     const float* __restrict__ gamma,
                                                     const float* __restrict__ beta,
                                                     float* __restrict__ out) {
    __shared__ float fin[2 * HH];
    __shared__ float red[256];
    const int b = blockIdx.x, t = threadIdx.x;
    for (int i = t; i < HH; i += 256) {
        fin[i]      = context[(size_t)b * HH + i];
        fin[HH + i] = qst[(size_t)b * HH + i];
    }
    __syncthreads();
    float acc[4];
#pragma unroll
    for (int i = 0; i < 4; ++i) {
        int h = t + (i << 8);
        acc[i] = bo[h] + qst[(size_t)b * HH + h];       // residual folded in
    }
    for (int k = 0; k < 2 * HH; ++k) {
        const float f = fin[k];
        const float* wr = Wo + (size_t)k * HH;
#pragma unroll
        for (int i = 0; i < 4; ++i) acc[i] = fmaf(f, wr[t + (i << 8)], acc[i]);
    }
    float lsum = 0.0f, lsq = 0.0f;
#pragma unroll
    for (int i = 0; i < 4; ++i) { lsum += acc[i]; lsq += acc[i] * acc[i]; }
    red[t] = lsum; __syncthreads();
    for (int o = 128; o > 0; o >>= 1) { if (t < o) red[t] += red[t + o]; __syncthreads(); }
    const float mu = red[0] / (float)HH;
    __syncthreads();
    red[t] = lsq; __syncthreads();
    for (int o = 128; o > 0; o >>= 1) { if (t < o) red[t] += red[t + o]; __syncthreads(); }
    const float var  = red[0] / (float)HH - mu * mu;
    const float rstd = rsqrtf(var + 1e-5f);
#pragma unroll
    for (int i = 0; i < 4; ++i) {
        int h = t + (i << 8);
        out[(size_t)b * HH + h] = (acc[i] - mu) * rstd * gamma[h] + beta[h];
    }
}

// ---------------------------------------------------------------------------
extern "C" void kernel_launch(void* const* d_in, const int* in_sizes, int n_in,
                              void* d_out, int out_size, void* d_ws, size_t ws_size,
                              hipStream_t stream) {
    (void)in_sizes; (void)n_in; (void)out_size; (void)ws_size;
    const float* seq   = (const float*)d_in[0];
    const float* qst   = (const float*)d_in[1];
    const float* Wk    = (const float*)d_in[2];
    const float* bk    = (const float*)d_in[3];
    const float* Wq    = (const float*)d_in[4];
    const float* bq    = (const float*)d_in[5];
    const float* Wg    = (const float*)d_in[6];
    const float* bg    = (const float*)d_in[7];
    const float* Ws    = (const float*)d_in[8];
    const float* bs    = (const float*)d_in[9];
    const float* Wo    = (const float*)d_in[10];
    const float* bo    = (const float*)d_in[11];
    const float* gamma = (const float*)d_in[12];
    const float* beta  = (const float*)d_in[13];
    float* out = (float*)d_out;

    char* ws = (char*)d_ws;
    _Float16* WkT    = (_Float16*)(ws);                               // 2 MB
    _Float16* WgT    = (_Float16*)(ws + (size_t)AA * HH * 2);         // 2 MB
    float*    query  = (float*)(ws + (size_t)AA * HH * 4);            // 128 KB
    float*    scores = (float*)((char*)query + (size_t)BB * AA * 4);  // 256 KB
    float*    ctx    = (float*)((char*)scores + (size_t)BB * SS * 4); // 128 KB

    conv_t_kernel<<<(HH * AA) / 256, 256, 0, stream>>>(Wk, WkT, HH, AA);
    conv_t_kernel<<<(AA * AA) / 256, 256, 0, stream>>>(Wg, WgT, AA, AA);
    qproj_kernel<<<(BB * AA) / 256, 256, 0, stream>>>(qst, Wq, bq, query);

    fused_attn_kernel<<<BB * (SS / SROWS), 256, 0, stream>>>(
        seq, query, WkT, bk, WgT, bg, Ws, bs, scores);

    float* weights = out + (size_t)BB * HH;   // second output region
    softmax_kernel<<<BB, 256, 0, stream>>>(scores, weights);
    context_kernel<<<BB * (HH / 256), 256, 0, stream>>>(weights, seq, ctx);
    out_ln_kernel<<<BB, 256, 0, stream>>>(ctx, qst, Wo, bo, gamma, beta, out);
}
